// ConditionGenerator_41197326303574
// MI455X (gfx1250) — matrix operand order repaired and checked by reading, not compiled, not run on hardware
//
#include <hip/hip_runtime.h>

typedef __attribute__((ext_vector_type(16))) _Float16 v16h;
typedef __attribute__((ext_vector_type(8)))  float    v8f;

// ---------------- problem constants ----------------
#define SRATE   16000
#define NB      4
#define NT      250
#define NS      64000
#define NH      101
#define NFFT    512
#define HOP     256
#define ND      257      // NFFT/2+1
#define NFR     251      // stft frames per batch
#define MROWS   1004     // NB*NFR
#define MPAD    1008     // 63*16
#define NPAD    272      // 17*16
#define KPAD    288      // 9*32  (for D=257 GEMMs)
#define KT_FFT  16       // 512/32
#define KT_MLP  9        // 288/32

// ---------------- WMMA fragment loaders (CDNA5 wave32 layouts) ----------------
// A fragment: 16x32 f16, row-major source with stride lda (in halfs).
// lanes 0-15: M=lane, K = 0..15; lanes 16-31: M=lane-16, K = 16..31
__device__ inline v16h load_a_frag(const _Float16* A, int lda, int mrow, int k0, int lane) {
    int m  = mrow + (lane & 15);
    int kb = k0 + ((lane >> 4) << 4);
    const _Float16* p = A + (size_t)m * lda + kb;
    v16h a;
#pragma unroll
    for (int e = 0; e < 8; ++e) a[e] = p[e];          // K = kb+0..7
#pragma unroll
    for (int e = 0; e < 8; ++e) a[8 + e] = p[8 + e]; // K = kb+8..15
    return a;
}

// B fragment: 32x16 f16 from fragment-friendly layout Bp[kt][n][klocal(32)].
// lane: N = lane&15; element e: K = e + 16*(lane>=16)
__device__ inline v16h load_b_frag(const _Float16* Bt, int ncol, int lane) {
    int n = ncol + (lane & 15);
    const _Float16* p = Bt + (size_t)n * 32 + ((lane >> 4) << 4);
    v16h b;
#pragma unroll
    for (int e = 0; e < 16; ++e) b[e] = p[e];
    return b;
}

// ---------------- small helper kernels ----------------
__global__ void zero_f16_kernel(_Float16* p, int n) {
    int i = blockIdx.x * blockDim.x + threadIdx.x;
    if (i < n) p[i] = (_Float16)0.f;
}

// harm_amps[b,t,k] = c * (c*mask) / max(c*count, 1e-7)
__global__ void harm_kernel(const float* f0, float* harm) {
    int gid = blockIdx.x * blockDim.x + threadIdx.x; // over NB*NT
    if (gid >= NB * NT) return;
    float f = f0[gid];
    int cnt = 0;
#pragma unroll 1
    for (int k = 1; k <= NH; ++k) cnt += (f * (float)k < 8000.f) ? 1 : 0;
    float s = 1.f / (1.f + expf(-1.f));
    float c = 2.f * powf(s, logf(10.f)) + 1e-7f;      // exp_sigmoid(1)
    float denom = fmaxf(c * (float)cnt, 1e-7f);
#pragma unroll 1
    for (int k = 1; k <= NH; ++k) {
        float hv = (f * (float)k < 8000.f) ? c : 0.f;
        harm[(size_t)gid * NH + (k - 1)] = c * (hv / denom);
    }
}

// per-batch inclusive scan of 2*pi*f0_up/SR  (one block of 256 per batch)
__global__ void phase_kernel(const float* f0, float* phase) {
    __shared__ float buf[256];
    __shared__ float carry;
    int b = blockIdx.x, tid = threadIdx.x;
    if (tid == 0) carry = 0.f;
    __syncthreads();
    const float cc = 2.f * 3.14159265358979323846f / (float)SRATE;
    for (int tile = 0; tile < NS / 256; ++tile) {
        int n = tile * 256 + tid;
        float t = (float)n * (249.0f / 63999.0f);
        int i0 = (int)floorf(t);
        int i1 = min(i0 + 1, NT - 1);
        float w = t - (float)i0;
        float f = f0[b * NT + i0] * (1.f - w) + f0[b * NT + i1] * w;
        buf[tid] = cc * f;
        __syncthreads();
#pragma unroll
        for (int off = 1; off < 256; off <<= 1) {
            float add = (tid >= off) ? buf[tid - off] : 0.f;
            __syncthreads();
            buf[tid] += add;
            __syncthreads();
        }
        phase[b * NS + n] = buf[tid] + carry;
        __syncthreads();
        if (tid == 255) carry = carry + buf[255];
        __syncthreads();
    }
}

// signal[b,n] = sum_k amps_up[b,n,k] * sin(phase*k)
// Chebyshev recurrence: sin((k+1)t) = 2cos(t) sin(kt) - sin((k-1)t)
// -> one sincosf per sample instead of 101 large-argument sinf calls.
__global__ void synth_kernel(const float* harm, const float* phase, float* signal) {
    int gid = blockIdx.x * blockDim.x + threadIdx.x; // NB*NS
    if (gid >= NB * NS) return;
    int b = gid / NS, n = gid % NS;
    float t = (float)n * (249.0f / 63999.0f);
    int i0 = (int)floorf(t);
    int i1 = min(i0 + 1, NT - 1);
    float w = t - (float)i0;
    float ph = phase[gid];
    const float* h0 = harm + (size_t)(b * NT + i0) * NH;
    const float* h1 = harm + (size_t)(b * NT + i1) * NH;
    float s1, c1;
    sincosf(ph, &s1, &c1);
    float twoc = 2.f * c1;
    float skm1 = 0.f, sk = s1;   // sin(0*ph), sin(1*ph)
    float acc = 0.f;
#pragma unroll 1
    for (int k = 0; k < NH; ++k) {
        float a = h0[k] * (1.f - w) + h1[k] * w;
        acc = fmaf(a, sk, acc);
        float skp1 = fmaf(twoc, sk, -skm1);
        skm1 = sk;
        sk = skp1;
    }
    signal[gid] = acc;
}

// windowed frames, f16, MPAD x NFFT (reflect padding)
__global__ void frames_kernel(const float* signal, _Float16* Afr) {
    int gid = blockIdx.x * blockDim.x + threadIdx.x; // MPAD*NFFT
    if (gid >= MPAD * NFFT) return;
    int m = gid >> 9, n = gid & (NFFT - 1);
    float v = 0.f;
    if (m < MROWS) {
        int b = m / NFR, f = m % NFR;
        int j = f * HOP + n - (NFFT / 2);
        if (j < 0) j = -j;
        if (j >= NS) j = 2 * (NS - 1) - j;
        float win = 0.5f - 0.5f * cospif((float)n / 256.0f); // cos(2*pi*n/512)
        v = signal[b * NS + j] * win;
    }
    Afr[gid] = (_Float16)v;
}

// DFT bases in fragment-friendly layout: B[kt][n][kl] , k = kt*32+kl in [0,512)
__global__ void basis_kernel(_Float16* Bcos, _Float16* Bsin) {
    int gid = blockIdx.x * blockDim.x + threadIdx.x; // KT_FFT*NPAD*32
    if (gid >= KT_FFT * NPAD * 32) return;
    int rem = gid % (NPAD * 32);
    int kt  = gid / (NPAD * 32);
    int n   = rem / 32;
    int kl  = rem % 32;
    int ks  = kt * 32 + kl;
    float cv = 0.f, sv = 0.f;
    if (n < ND) {
        float x = (float)(n * ks) / 256.0f; // pi*x = 2*pi*n*ks/512
        cv = cospif(x);
        sv = sinpif(x);
    }
    Bcos[gid] = (_Float16)cv;
    Bsin[gid] = (_Float16)sv;
}

// weight (D x D, row-major) -> fragment layout Bp[kt][n][kl] = W[n][k]  (W.T GEMM)
__global__ void wprep_kernel(const float* W, _Float16* Bp) {
    int gid = blockIdx.x * blockDim.x + threadIdx.x; // KT_MLP*NPAD*32
    if (gid >= KT_MLP * NPAD * 32) return;
    int rem = gid % (NPAD * 32);
    int kt  = gid / (NPAD * 32);
    int n   = rem / 32;
    int kl  = rem % 32;
    int k   = kt * 32 + kl;
    float v = (k < ND && n < ND) ? W[n * ND + k] : 0.f;
    Bp[gid] = (_Float16)v;
}

// ---------------- WMMA GEMM kernels (1 wave = one 16x16 tile) ----------------
// STFT magnitude: mag = sqrt((F@cos)^2 + (F@sin)^2), stored f16 into magA (stride KPAD)
__global__ void gemm_fft_kernel(const _Float16* Afr, const _Float16* Bcos,
                                const _Float16* Bsin, _Float16* magA) {
    int tm = blockIdx.x, tn = blockIdx.y, lane = threadIdx.x;
    v8f cre = {}, cim = {};
#pragma unroll 1
    for (int kt = 0; kt < KT_FFT; ++kt) {
        v16h a  = load_a_frag(Afr, NFFT, tm * 16, kt * 32, lane);
        v16h bc = load_b_frag(Bcos + (size_t)kt * NPAD * 32, tn * 16, lane);
        v16h bs = load_b_frag(Bsin + (size_t)kt * NPAD * 32, tn * 16, lane);
        cre = __builtin_amdgcn_wmma_f32_16x16x32_f16(false, a, false, bc, (short)0, cre, false, false);
        cim = __builtin_amdgcn_wmma_f32_16x16x32_f16(false, a, false, bs, (short)0, cim, false, false);
    }
    int n  = tn * 16 + (lane & 15);
    int m0 = tm * 16 + ((lane >> 4) << 3);
#pragma unroll
    for (int r = 0; r < 8; ++r) {
        float re = cre[r], im = cim[r];
        magA[(size_t)(m0 + r) * KPAD + n] = (_Float16)sqrtf(re * re + im * im);
    }
}

// MLP layer 1 (both chains via blockIdx.z): H = tanh(magA @ W1.T + b1), f16 store
__global__ void gemm_mlp1_kernel(const _Float16* A,
                                 const _Float16* Bg, const float* biasg, _Float16* Hg,
                                 const _Float16* Bb, const float* biasb, _Float16* Hb) {
    int tm = blockIdx.x, tn = blockIdx.y, lane = threadIdx.x;
    const _Float16* Bp   = blockIdx.z ? Bb : Bg;
    const float*    bias = blockIdx.z ? biasb : biasg;
    _Float16*       Hout = blockIdx.z ? Hb : Hg;
    v8f c = {};
#pragma unroll 1
    for (int kt = 0; kt < KT_MLP; ++kt) {
        v16h a = load_a_frag(A, KPAD, tm * 16, kt * 32, lane);
        v16h b = load_b_frag(Bp + (size_t)kt * NPAD * 32, tn * 16, lane);
        c = __builtin_amdgcn_wmma_f32_16x16x32_f16(false, a, false, b, (short)0, c, false, false);
    }
    int n  = tn * 16 + (lane & 15);
    int m0 = tm * 16 + ((lane >> 4) << 3);
    float bv = (n < ND) ? bias[n] : 0.f;
#pragma unroll
    for (int r = 0; r < 8; ++r)
        Hout[(size_t)(m0 + r) * KPAD + n] = (_Float16)tanhf(c[r] + bv);
}

// MLP layer 2 (both chains via blockIdx.z): O = relu(H @ W2.T + b2),
// stored f32 transposed into (b, d, t) layout.
__global__ void gemm_mlp2_kernel(const _Float16* Hg, const _Float16* Bg, const float* biasg, float* Og,
                                 const _Float16* Hb, const _Float16* Bb, const float* biasb, float* Ob) {
    int tm = blockIdx.x, tn = blockIdx.y, lane = threadIdx.x;
    const _Float16* A    = blockIdx.z ? Hb : Hg;
    const _Float16* Bp   = blockIdx.z ? Bb : Bg;
    const float*    bias = blockIdx.z ? biasb : biasg;
    float*          Oout = blockIdx.z ? Ob : Og;
    v8f c = {};
#pragma unroll 1
    for (int kt = 0; kt < KT_MLP; ++kt) {
        v16h a = load_a_frag(A, KPAD, tm * 16, kt * 32, lane);
        v16h b = load_b_frag(Bp + (size_t)kt * NPAD * 32, tn * 16, lane);
        c = __builtin_amdgcn_wmma_f32_16x16x32_f16(false, a, false, b, (short)0, c, false, false);
    }
    int n  = tn * 16 + (lane & 15);
    int m0 = tm * 16 + ((lane >> 4) << 3);
    if (n >= ND) return;
    float bv = bias[n];
#pragma unroll
    for (int r = 0; r < 8; ++r) {
        int m = m0 + r;
        if (m < MROWS) {
            int b = m / NFR, t = m % NFR;
            Oout[((size_t)b * ND + n) * NFR + t] = fmaxf(c[r] + bv, 0.f);
        }
    }
}

// ---------------- launcher ----------------
extern "C" void kernel_launch(void* const* d_in, const int* in_sizes, int n_in,
                              void* d_out, int out_size, void* d_ws, size_t ws_size,
                              hipStream_t stream) {
    (void)in_sizes; (void)n_in; (void)out_size; (void)ws_size;
    const float* f0  = (const float*)d_in[0];
    const float* wg1 = (const float*)d_in[1];
    const float* bg1 = (const float*)d_in[2];
    const float* wg2 = (const float*)d_in[3];
    const float* bg2 = (const float*)d_in[4];
    const float* wb1 = (const float*)d_in[5];
    const float* bb1 = (const float*)d_in[6];
    const float* wb2 = (const float*)d_in[7];
    const float* bb2 = (const float*)d_in[8];
    float* out = (float*)d_out;
    float* out_beta  = out;                                   // (4,257,251)
    float* out_gamma = out + (size_t)NB * ND * NFR;           // (4,257,251)

    // workspace carve-up
    char* base = (char*)d_ws;
    size_t off = 0;
    auto alloc = [&](size_t bytes) { off = (off + 255) & ~(size_t)255; size_t o = off; off += bytes; return o; };
    float*    harm   = (float*)   (base + alloc((size_t)NB * NT * NH * 4));
    float*    phase  = (float*)   (base + alloc((size_t)NB * NS * 4));
    float*    signal = (float*)   (base + alloc((size_t)NB * NS * 4));
    _Float16* Afr    = (_Float16*)(base + alloc((size_t)MPAD * NFFT * 2));
    _Float16* Bcos   = (_Float16*)(base + alloc((size_t)KT_FFT * NPAD * 32 * 2));
    _Float16* Bsin   = (_Float16*)(base + alloc((size_t)KT_FFT * NPAD * 32 * 2));
    _Float16* W1g    = (_Float16*)(base + alloc((size_t)KT_MLP * NPAD * 32 * 2));
    _Float16* W2g    = (_Float16*)(base + alloc((size_t)KT_MLP * NPAD * 32 * 2));
    _Float16* W1b    = (_Float16*)(base + alloc((size_t)KT_MLP * NPAD * 32 * 2));
    _Float16* W2b    = (_Float16*)(base + alloc((size_t)KT_MLP * NPAD * 32 * 2));
    _Float16* magA   = (_Float16*)(base + alloc((size_t)MPAD * KPAD * 2));
    _Float16* Hg     = (_Float16*)(base + alloc((size_t)MPAD * KPAD * 2));
    _Float16* Hb     = (_Float16*)(base + alloc((size_t)MPAD * KPAD * 2));

    const int BS = 256;
    // zero padded f16 GEMM inputs (poison safety in padded K columns)
    zero_f16_kernel<<<(MPAD * KPAD + BS - 1) / BS, BS, 0, stream>>>(magA, MPAD * KPAD);
    zero_f16_kernel<<<(MPAD * KPAD + BS - 1) / BS, BS, 0, stream>>>(Hg, MPAD * KPAD);
    zero_f16_kernel<<<(MPAD * KPAD + BS - 1) / BS, BS, 0, stream>>>(Hb, MPAD * KPAD);

    harm_kernel <<<(NB * NT + BS - 1) / BS, BS, 0, stream>>>(f0, harm);
    phase_kernel<<<NB, 256, 0, stream>>>(f0, phase);
    synth_kernel<<<(NB * NS) / BS, BS, 0, stream>>>(harm, phase, signal);
    frames_kernel<<<(MPAD * NFFT) / BS, BS, 0, stream>>>(signal, Afr);
    basis_kernel<<<(KT_FFT * NPAD * 32 + BS - 1) / BS, BS, 0, stream>>>(Bcos, Bsin);
    wprep_kernel<<<(KT_MLP * NPAD * 32 + BS - 1) / BS, BS, 0, stream>>>(wg1, W1g);
    wprep_kernel<<<(KT_MLP * NPAD * 32 + BS - 1) / BS, BS, 0, stream>>>(wg2, W2g);
    wprep_kernel<<<(KT_MLP * NPAD * 32 + BS - 1) / BS, BS, 0, stream>>>(wb1, W1b);
    wprep_kernel<<<(KT_MLP * NPAD * 32 + BS - 1) / BS, BS, 0, stream>>>(wb2, W2b);

    dim3 gemm_grid(MPAD / 16, NPAD / 16);        // 63 x 17, one wave per tile
    dim3 gemm_grid2(MPAD / 16, NPAD / 16, 2);    // both MLP chains concurrently
    gemm_fft_kernel<<<gemm_grid, 32, 0, stream>>>(Afr, Bcos, Bsin, magA);
    gemm_mlp1_kernel<<<gemm_grid2, 32, 0, stream>>>(magA, W1g, bg1, Hg, W1b, bb1, Hb);
    gemm_mlp2_kernel<<<gemm_grid2, 32, 0, stream>>>(Hg, W2g, bg2, out_gamma, Hb, W2b, bb2, out_beta);
}
